// TIMITModelPytorchCRF_12120397710175
// MI455X (gfx1250) — compile-verified, matrix operand order checked by reading, not run on hardware
//
#include <hip/hip_runtime.h>

// ---------------- problem constants ----------------
#define BB   64
#define TT   800
#define DD   80
#define HH   128
#define KK   39
#define BT   (BB*TT)       // 51200
#define G4   (4*HH)        // 512
#define KP1  96            // D padded to multiple of 32
#define KP3  (2*HH)        // 256
#define NPAD 48            // K padded to multiple of 16

typedef __attribute__((ext_vector_type(16))) __bf16 v16bf;
typedef __attribute__((ext_vector_type(8)))  __bf16 v8bf;
typedef __attribute__((ext_vector_type(8)))  float  v8f;

union U16 { v16bf v; v8bf h[2]; };

// ---------------- staging kernels ----------------
__global__ void k_cvt_pad(const float* __restrict__ src, __bf16* __restrict__ dst,
                          int rin, int rout, int kin, int kout) {
  long n = (long)rout * kout;
  long i = (long)blockIdx.x * blockDim.x + threadIdx.x;
  if (i >= n) return;
  int r = (int)(i / kout), k = (int)(i % kout);
  float v = (r < rin && k < kin) ? src[(long)r * kin + k] : 0.0f;
  dst[i] = (__bf16)v;
}

__global__ void k_addvec(const float* __restrict__ a, const float* __restrict__ b,
                         float* __restrict__ dst, int n) {
  int i = blockIdx.x * blockDim.x + threadIdx.x;
  if (i < n) dst[i] = a[i] + b[i];
}

// ---------------- bf16 WMMA GEMM, NBT N-tiles per wave (A-fragment reuse) ----------------
// C(MxN) = A(MxKp) * Bm(NxKp)^T + bias
// TSWZ: interpret A-row as (b*TT + t) and store C row at (t*BB + b)  (timestep-major)
template <int NBT, bool TSWZ>
__global__ __launch_bounds__(128) void k_gemm_bf16(
    const __bf16* __restrict__ A,    // M x Kp, row-major
    const __bf16* __restrict__ Bm,   // N x Kp, row-major (pre-transposed)
    const float*  __restrict__ bias, // N
    float*        __restrict__ C,    // M x ldC
    int M, int N, int Kp, int ldC, int Nvalid) {
  int wave = blockIdx.x * (blockDim.x >> 5) + (threadIdx.x >> 5);
  int lane = threadIdx.x & 31;
  int ngrps = N / (16 * NBT);
  int mtile = wave / ngrps, ngrp = wave % ngrps;
  if (mtile * 16 >= M) return;
  int l15 = lane & 15;
  int hi  = (lane >> 4) ? 8 : 0;
  long rowA = (long)(mtile * 16 + l15) * Kp;
  long rowB[NBT];
#pragma unroll
  for (int q = 0; q < NBT; ++q)
    rowB[q] = (long)((ngrp * NBT + q) * 16 + l15) * Kp;

  v8f c[NBT];
#pragma unroll
  for (int q = 0; q < NBT; ++q)
    for (int r = 0; r < 8; ++r) c[q][r] = 0.0f;

  for (int k0 = 0; k0 < Kp; k0 += 32) {
    int kb = k0 + hi;
    U16 ua, ub[NBT];
    ua.h[0] = *(const v8bf*)(A + rowA + kb);
    ua.h[1] = *(const v8bf*)(A + rowA + kb + 16);
#pragma unroll
    for (int q = 0; q < NBT; ++q) {
      ub[q].h[0] = *(const v8bf*)(Bm + rowB[q] + kb);
      ub[q].h[1] = *(const v8bf*)(Bm + rowB[q] + kb + 16);
    }
#pragma unroll
    for (int q = 0; q < NBT; ++q)
      c[q] = __builtin_amdgcn_wmma_f32_16x16x32_bf16(false, ua.v, false, ub[q].v,
                                                     (short)0, c[q], false, false);
  }
#pragma unroll
  for (int q = 0; q < NBT; ++q) {
    int col = (ngrp * NBT + q) * 16 + l15;
    if (col < Nvalid) {
      float bv = bias ? bias[col] : 0.0f;
      for (int r = 0; r < 8; ++r) {
        int row = mtile * 16 + r + hi;
        long orow;
        if (TSWZ) {
          int bidx = row / TT, tidx = row % TT;   // row == b*TT + t
          orow = (long)tidx * BB + bidx;          // timestep-major
        } else {
          orow = row;
        }
        C[orow * ldC + col] = c[q][r] + bv;
      }
    }
  }
}

// ---------------- persistent LDS-resident bidirectional LSTM recurrence ----------------
// grid = 2 blocks (dir 0 = fwd, dir 1 = bwd), 1024 threads = 32 waves, one WGP each.
__global__ __launch_bounds__(1024) void k_lstm(
    const __bf16* __restrict__ WhhSt,    // 2 x 512 x 128 bf16 (gate-major)
    const float*  __restrict__ xW,       // 2 x TT x BB x 512 f32 (timestep-major)
    __bf16*       __restrict__ hiddenBf) // TT x BB x 256 bf16 (timestep-major)
{
  __shared__ __bf16 Wlds[G4 * HH];    // 128 KB
  __shared__ float  Glds[BB * G4];    // 128 KB
  __shared__ float  clds[BB * HH];    //  32 KB
  __shared__ __bf16 hlds[BB * HH];    //  16 KB  (304 KB total)

  const int dir  = blockIdx.x;
  const int tid  = threadIdx.x;
  const int wave = tid >> 5, lane = tid & 31;
  const int l15  = lane & 15;
  const int hi   = (lane >> 4) ? 8 : 0;

  // ---- async fill of Whh into LDS (CDNA5 GLOBAL_LOAD_ASYNC_TO_LDS_B128) ----
  {
    const char* Wg = (const char*)(WhhSt + (long)dir * G4 * HH);
    unsigned ldsBase = (unsigned)(size_t)(void*)Wlds;  // low 32b of generic LDS ptr = LDS offset
    for (int i = tid; i < 8192; i += 1024) {           // 128 KB = 8192 x 16B
      unsigned ldsOff = ldsBase + i * 16u;
      const char* gp = Wg + (long)i * 16;
      asm volatile("global_load_async_to_lds_b128 %0, %1, off"
                   :: "v"(ldsOff), "v"(gp) : "memory");
    }
    asm volatile("s_wait_asynccnt 0" ::: "memory");
  }
  for (int i = tid; i < BB * HH; i += 1024) { clds[i] = 0.0f; hlds[i] = (__bf16)0.0f; }
  __syncthreads();

  const float* xwd = xW + (long)dir * TT * BB * G4;

  // wave -> one M tile x 4 consecutive N tiles (A fragment reused 4x per k-chunk)
  const int mtile = wave & 3;        // 4 M tiles (64 batch rows)
  const int ngrp  = wave >> 2;       // 8 groups x 4 N tiles = 512 cols
  const int rowA  = (mtile * 16 + l15) * HH;
  int rowB[4];
#pragma unroll
  for (int q = 0; q < 4; ++q) rowB[q] = ((ngrp * 4 + q) * 16 + l15) * HH;

  for (int tt = 0; tt < TT; ++tt) {
    const int t = dir ? (TT - 1 - tt) : tt;
    // ---- Phase A: G = xW[t] + h @ Whh^T ----
    const float* xwt = xwd + (long)t * BB * G4;   // contiguous 128 KB block for step t
    v8f c[4];
#pragma unroll
    for (int q = 0; q < 4; ++q) {
      const int col = (ngrp * 4 + q) * 16 + l15;
      for (int r = 0; r < 8; ++r) {
        int b = mtile * 16 + r + hi;
        c[q][r] = xwt[(long)b * G4 + col];
      }
    }
    // prefetch next step's seed block while this step computes (global_prefetch_b8)
    if (tt + 1 < TT) {
      const int tn = dir ? (TT - 2 - tt) : tt + 1;
      __builtin_prefetch(xwd + (long)tn * BB * G4 + tid * 32, 0, 0);
    }
#pragma unroll
    for (int k0 = 0; k0 < HH; k0 += 32) {
      int kb = k0 + hi;
      U16 ua, ub0, ub1, ub2, ub3;
      ua.h[0]  = *(const v8bf*)(hlds + rowA + kb);
      ua.h[1]  = *(const v8bf*)(hlds + rowA + kb + 16);
      ub0.h[0] = *(const v8bf*)(Wlds + rowB[0] + kb);
      ub0.h[1] = *(const v8bf*)(Wlds + rowB[0] + kb + 16);
      ub1.h[0] = *(const v8bf*)(Wlds + rowB[1] + kb);
      ub1.h[1] = *(const v8bf*)(Wlds + rowB[1] + kb + 16);
      ub2.h[0] = *(const v8bf*)(Wlds + rowB[2] + kb);
      ub2.h[1] = *(const v8bf*)(Wlds + rowB[2] + kb + 16);
      ub3.h[0] = *(const v8bf*)(Wlds + rowB[3] + kb);
      ub3.h[1] = *(const v8bf*)(Wlds + rowB[3] + kb + 16);
      c[0] = __builtin_amdgcn_wmma_f32_16x16x32_bf16(false, ua.v, false, ub0.v,
                                                     (short)0, c[0], false, false);
      c[1] = __builtin_amdgcn_wmma_f32_16x16x32_bf16(false, ua.v, false, ub1.v,
                                                     (short)0, c[1], false, false);
      c[2] = __builtin_amdgcn_wmma_f32_16x16x32_bf16(false, ua.v, false, ub2.v,
                                                     (short)0, c[2], false, false);
      c[3] = __builtin_amdgcn_wmma_f32_16x16x32_bf16(false, ua.v, false, ub3.v,
                                                     (short)0, c[3], false, false);
    }
#pragma unroll
    for (int q = 0; q < 4; ++q) {
      const int col = (ngrp * 4 + q) * 16 + l15;
      for (int r = 0; r < 8; ++r) {
        int b = mtile * 16 + r + hi;
        Glds[b * G4 + col] = c[q][r];
      }
    }
    __syncthreads();
    // ---- Phase B: gate activations, state update (coalesced timestep-major store) ----
    __bf16* hout = hiddenBf + (long)t * BB * KP3 + dir * HH;
    for (int e = tid; e < BB * HH; e += 1024) {
      int b = e >> 7, j = e & (HH - 1);
      float gi = Glds[b * G4 + j];
      float gf = Glds[b * G4 + HH + j];
      float gg = Glds[b * G4 + 2 * HH + j];
      float go = Glds[b * G4 + 3 * HH + j];
      float si = 1.0f / (1.0f + __expf(-gi));
      float sf = 1.0f / (1.0f + __expf(-gf));
      float so = 1.0f / (1.0f + __expf(-go));
      float cn = sf * clds[e] + si * tanhf(gg);
      float hn = so * tanhf(cn);
      clds[e] = cn;
      hlds[e] = (__bf16)hn;
      hout[(long)b * KP3 + j] = (__bf16)hn;
    }
    __syncthreads();
  }
}

// ---------------- CRF NLL: one 64-thread block per sequence ----------------
// emissions are timestep-major: em[(t*BB + b)*KK + k]
__global__ __launch_bounds__(64) void k_crf(
    const float* __restrict__ em,
    const int*   __restrict__ labels,  // B x T
    const int*   __restrict__ lengths, // B
    const float* __restrict__ start_t, const float* __restrict__ end_t,
    const float* __restrict__ trans,   // 39 x 39
    float* __restrict__ out) {
  __shared__ float tr[KK * KK];
  __shared__ float alpha[KK];
  __shared__ float red[64];
  const int b = blockIdx.x, tid = threadIdx.x;
  const long EMR = (long)BB * KK;      // per-timestep emission row stride
  for (int i = tid; i < KK * KK; i += 64) tr[i] = trans[i];
  const int len = lengths[b];
  const float* emb = em + (long)b * KK;
  const int*   lab = labels + (long)b * TT;

  float np = 0.0f;
  for (int t = 1 + tid; t < len; t += 64)
    np += trans[lab[t - 1] * KK + lab[t]] + emb[t * EMR + lab[t]];
  red[tid] = np;
  __syncthreads();

  if (tid < KK) alpha[tid] = start_t[tid] + emb[tid];
  __syncthreads();

  for (int t = 1; t < len; ++t) {
    float v = 0.0f;
    if (tid < KK) {
      float m = -1e30f;
      for (int i = 0; i < KK; ++i) m = fmaxf(m, alpha[i] + tr[i * KK + tid]);
      float s = 0.0f;
      for (int i = 0; i < KK; ++i) s += __expf(alpha[i] + tr[i * KK + tid] - m);
      v = m + __logf(s) + emb[t * EMR + tid];
    }
    __syncthreads();
    if (tid < KK) alpha[tid] = v;
    __syncthreads();
  }

  if (tid == 0) {
    float num = start_t[lab[0]] + emb[lab[0]];
    for (int i = 0; i < 64; ++i) num += red[i];
    num += end_t[lab[len - 1]];
    float m = -1e30f;
    for (int j = 0; j < KK; ++j) m = fmaxf(m, alpha[j] + end_t[j]);
    float s = 0.0f;
    for (int j = 0; j < KK; ++j) s += __expf(alpha[j] + end_t[j] - m);
    float denom = m + __logf(s);
    atomicAdd(out, (denom - num) * (1.0f / (float)BB));
  }
}

// ---------------- host launcher ----------------
extern "C" void kernel_launch(void* const* d_in, const int* in_sizes, int n_in,
                              void* d_out, int out_size, void* d_ws, size_t ws_size,
                              hipStream_t stream) {
  const float* features = (const float*)d_in[0];
  const int*   lengths  = (const int*)  d_in[1];
  const int*   labels   = (const int*)  d_in[2];
  const float* Wih_f    = (const float*)d_in[3];
  const float* Whh_f    = (const float*)d_in[4];
  const float* bih_f    = (const float*)d_in[5];
  const float* bhh_f    = (const float*)d_in[6];
  const float* Wih_b    = (const float*)d_in[7];
  const float* Whh_b    = (const float*)d_in[8];
  const float* bih_b    = (const float*)d_in[9];
  const float* bhh_b    = (const float*)d_in[10];
  const float* Wp       = (const float*)d_in[11];
  const float* bp       = (const float*)d_in[12];
  const float* start_t  = (const float*)d_in[13];
  const float* end_t    = (const float*)d_in[14];
  const float* trans    = (const float*)d_in[15];

  char* ws = (char*)d_ws;
  __bf16* Xbf     = (__bf16*)ws;                 ws += (size_t)BT * KP1 * 2;
  __bf16* WihSt   = (__bf16*)ws;                 ws += (size_t)2 * G4 * KP1 * 2;
  __bf16* WhhSt   = (__bf16*)ws;                 ws += (size_t)2 * G4 * HH * 2;
  float*  biasSt  = (float*)ws;                  ws += (size_t)2 * G4 * 4;
  __bf16* WpSt    = (__bf16*)ws;                 ws += (size_t)NPAD * KP3 * 2;
  float*  xW      = (float*)ws;                  ws += (size_t)2 * BT * G4 * 4;
  __bf16* hidBf   = (__bf16*)ws;                 ws += (size_t)BT * KP3 * 2;
  float*  emis    = (float*)ws;

  {
    long n = (long)BT * KP1;
    k_cvt_pad<<<(unsigned)((n + 255) / 256), 256, 0, stream>>>(features, Xbf, BT, BT, DD, KP1);
  }
  k_cvt_pad<<<(G4 * KP1 + 255) / 256, 256, 0, stream>>>(Wih_f, WihSt,            G4, G4, DD, KP1);
  k_cvt_pad<<<(G4 * KP1 + 255) / 256, 256, 0, stream>>>(Wih_b, WihSt + G4 * KP1, G4, G4, DD, KP1);
  k_cvt_pad<<<(G4 * HH  + 255) / 256, 256, 0, stream>>>(Whh_f, WhhSt,            G4, G4, HH, HH);
  k_cvt_pad<<<(G4 * HH  + 255) / 256, 256, 0, stream>>>(Whh_b, WhhSt + G4 * HH,  G4, G4, HH, HH);
  k_cvt_pad<<<(NPAD * KP3 + 255) / 256, 256, 0, stream>>>(Wp, WpSt, KK, NPAD, KP3, KP3);
  k_addvec<<<2, 512, 0, stream>>>(bih_f, bhh_f, biasSt,      G4);
  k_addvec<<<2, 512, 0, stream>>>(bih_b, bhh_b, biasSt + G4, G4);

  // input projection GEMMs (timestep-major output): 3200 M-tiles x 8 N-groups
  {
    int waves  = (BT / 16) * (G4 / 64);   // 25600
    int blocks = waves / 4;
    k_gemm_bf16<4, true><<<blocks, 128, 0, stream>>>(Xbf, WihSt, biasSt,
                                                     xW, BT, G4, KP1, G4, G4);
    k_gemm_bf16<4, true><<<blocks, 128, 0, stream>>>(Xbf, WihSt + G4 * KP1, biasSt + G4,
                                                     xW + (size_t)BT * G4, BT, G4, KP1, G4, G4);
  }

  k_lstm<<<2, 1024, 0, stream>>>(WhhSt, xW, hidBf);

  // projection GEMM: hidden and emissions both timestep-major, rows map 1:1
  {
    int waves = (BT / 16);                // 3200
    k_gemm_bf16<3, false><<<waves / 4, 128, 0, stream>>>(hidBf, WpSt, bp,
                                                         emis, BT, NPAD, KP3, KK, KK);
  }

  hipMemsetAsync(d_out, 0, sizeof(float), stream);
  k_crf<<<BB, 64, 0, stream>>>(emis, labels, lengths, start_t, end_t, trans, (float*)d_out);
}
